// Block_31387620999284
// MI455X (gfx1250) — matrix operand order repaired and checked by reading, not compiled, not run on hardware
//
#include <hip/hip_runtime.h>

typedef __attribute__((ext_vector_type(16))) _Float16 v16h;
typedef __attribute__((ext_vector_type(8)))  float    v8f;
typedef unsigned int tdm_u32x4 __attribute__((ext_vector_type(4)));
typedef int          tdm_i32x8 __attribute__((ext_vector_type(8)));
typedef int          tdm_i32x4 __attribute__((ext_vector_type(4)));

union AFrag { v16h v; _Float16 e[16]; };
union CFrag { v8f  v; float    e[8];  };

#define LOG2E 1.44269504088896f

// ---------------------------------------------------------------------------
// Tensor Data Mover: 2D tile load Global -> LDS with row padding (ISA ch.8)
// tile_d0 elements (f16) per row, tile_d1 rows, row stride stride0 elements.
// LDS receives rows padded: pad_interval_code (0=2,1=4,2=8,3=16,4=32 DWORDs),
// pad_amount_code (+1 DWORDs of pad).
// ---------------------------------------------------------------------------
__device__ __forceinline__ void tdm_load_2d(const void* gaddr, unsigned lds_off,
                                            unsigned tile_d0, unsigned tile_d1,
                                            unsigned stride0,
                                            unsigned pad_interval_code,
                                            unsigned pad_amount_code) {
    unsigned long long ga = (unsigned long long)gaddr;
    tdm_u32x4 g0;
    g0[0] = 1u;                                             // count=1, user mode
    g0[1] = lds_off;                                        // lds_addr (bytes)
    g0[2] = (unsigned)ga;                                   // global_addr[31:0]
    g0[3] = (unsigned)((ga >> 32) & 0x01FFFFFFu) | (2u << 30); // addr[56:32] | type=2
    tdm_i32x8 g1;
    // data_size=1 (2B) | pad_enable | pad_interval | pad_amount ; wg_mask=0
    g1[0] = (int)((1u << 16) | (1u << 20) |
                  (pad_interval_code << 22) | (pad_amount_code << 25));
    // tensor_dim0 = tensor_dim1 = 2^30 (never OOB for our in-bounds tiles)
    g1[1] = 0;                                              // td0[15:0]<<16 = 0
    g1[2] = 0x4000;                                         // td0[31:16] | td1[15:0]<<16
    g1[3] = (int)(0x4000u | (tile_d0 << 16));               // td1[31:16] | tile_dim0
    g1[4] = (int)(tile_d1 & 0xFFFFu);                       // tile_dim1 | tile_dim2=0
    g1[5] = (int)stride0;                                   // tensor_dim0_stride[31:0]
    g1[6] = 0;                                              // stride0[47:32] | stride1[15:0]
    g1[7] = 0;                                              // stride1[47:16]
    tdm_i32x4 gz4 = {0, 0, 0, 0};
    tdm_i32x8 gz8 = {0, 0, 0, 0, 0, 0, 0, 0};
    __builtin_amdgcn_tensor_load_to_lds(g0, g1, gz4, gz4, gz8, 0);
}

// ---------------------------------------------------------------------------
// WMMA fragment loaders (CDNA5 wave32 layouts, ISA 7.12.2)
// ---------------------------------------------------------------------------

// A-matrix 16x32 f16. base -> element [m=0][k=0], ld = row stride in halves.
__device__ __forceinline__ v16h load_frag_a(const _Float16* base, int ld) {
    int lane = threadIdx.x & 31;
    int m  = lane & 15;
    int kb = (lane >> 4) << 3;       // 0 or 8
    const _Float16* r = base + m * ld + kb;
    AFrag f;
#pragma unroll
    for (int i = 0; i < 8; ++i) { f.e[i] = r[i]; f.e[8 + i] = r[16 + i]; }
    return f.v;
}

// B-matrix 32x16 f16 where LDS holds the TRANSPOSE: base -> [n=0][k=0].
__device__ __forceinline__ v16h load_frag_bT(const _Float16* base, int ld) {
    int lane = threadIdx.x & 31;
    int n  = lane & 15;
    int kb = (lane >> 4) << 3;       // 0 or 8
    const _Float16* r = base + n * ld + kb;
    AFrag f;
#pragma unroll
    for (int i = 0; i < 8; ++i) { f.e[i] = r[i]; f.e[8 + i] = r[16 + i]; }
    return f.v;
}

// ---------------------------------------------------------------------------
// VALU xor-shuffle reductions within 16-lane rows via v_permlane16_b32
// ---------------------------------------------------------------------------
__device__ __forceinline__ float pl16(float v, unsigned lo, unsigned hi) {
    int r = __builtin_amdgcn_permlane16(__float_as_int(v), __float_as_int(v),
                                        (int)lo, (int)hi, false, false);
    return __int_as_float(r);
}
__device__ __forceinline__ float rowmax16(float v) {
    v = fmaxf(v, pl16(v, 0x67452301u, 0xEFCDAB89u));   // xor 1
    v = fmaxf(v, pl16(v, 0x54761032u, 0xDCFE98BAu));   // xor 2
    v = fmaxf(v, pl16(v, 0x32107654u, 0xBA98FEDCu));   // xor 4
    v = fmaxf(v, pl16(v, 0xFEDCBA98u, 0x76543210u));   // xor 8
    return v;
}
__device__ __forceinline__ float rowsum16(float v) {
    v += pl16(v, 0x67452301u, 0xEFCDAB89u);
    v += pl16(v, 0x54761032u, 0xDCFE98BAu);
    v += pl16(v, 0x32107654u, 0xBA98FEDCu);
    v += pl16(v, 0xFEDCBA98u, 0x76543210u);
    return v;
}

// raw v_exp_f32: softmax args are always <= 0, no libm range fixup needed
__device__ __forceinline__ float fast_exp(float x) {
    return __builtin_amdgcn_exp2f(x * LOG2E);
}

// ---------------------------------------------------------------------------
// f32 -> f16 convert
// ---------------------------------------------------------------------------
__global__ void cvt_kernel(const float* __restrict__ src, _Float16* __restrict__ dst, int n) {
    int i = blockIdx.x * 256 + threadIdx.x;
    if (i < n) dst[i] = (_Float16)src[i];
}

// ---------------------------------------------------------------------------
// LayerNorm (one row of D per block) -> f16 output
// ---------------------------------------------------------------------------
__global__ __launch_bounds__(256)
void ln_kernel(const float* __restrict__ x, const float* __restrict__ g,
               const float* __restrict__ bta, _Float16* __restrict__ out, int Dd) {
    __shared__ float s1[256], s2[256];
    int row = blockIdx.x;
    const float* xr = x + (size_t)row * Dd;
    float a = 0.f, b = 0.f;
    for (int i = threadIdx.x; i < Dd; i += 256) { float v = xr[i]; a += v; b += v * v; }
    s1[threadIdx.x] = a; s2[threadIdx.x] = b;
    __syncthreads();
    for (int st = 128; st > 0; st >>= 1) {
        if ((int)threadIdx.x < st) {
            s1[threadIdx.x] += s1[threadIdx.x + st];
            s2[threadIdx.x] += s2[threadIdx.x + st];
        }
        __syncthreads();
    }
    float mu  = s1[0] / (float)Dd;
    float var = s2[0] / (float)Dd - mu * mu;
    float rs  = rsqrtf(var + 1e-5f);
    _Float16* orow = out + (size_t)row * Dd;
    for (int i = threadIdx.x; i < Dd; i += 256)
        orow[i] = (_Float16)(((xr[i] - mu) * rs) * g[i] + bta[i]);
}

// ---------------------------------------------------------------------------
// Tiled f16 WMMA GEMM: C[M,N] = A[M,K] @ B[K,N] (+bias)(+resid)(relu)
// Block tile 128x128, 8 waves. A tile staged by the Tensor Data Mover
// (async DMA, overlapped with manual transposed B staging).
// ---------------------------------------------------------------------------
template <bool RELU, bool RESID, bool OUTF16>
__global__ __launch_bounds__(256)
void gemm_f16_kernel(const _Float16* __restrict__ A, const _Float16* __restrict__ B,
                     const float* __restrict__ bias, const float* __restrict__ resid,
                     void* __restrict__ out, int M, int N, int K) {
    __shared__ __align__(16) _Float16 As[128 * 40];   // 128 x 32 (m-major), stride 40
    __shared__ __align__(16) _Float16 Bs[128 * 40];   // 128 x 32 (n-major = B^T), stride 40
    const int tid  = threadIdx.x;
    const int lane = tid & 31;
    const int wave = tid >> 5;
    const int wm = wave & 3;
    const int wn = wave >> 2;
    const int bm = blockIdx.y * 128;
    const int bn = blockIdx.x * 128;
    const unsigned As_lds = (unsigned)(size_t)&As[0];  // LDS byte offset (addr[31:0])

    CFrag acc[2][4];
#pragma unroll
    for (int i = 0; i < 2; ++i)
#pragma unroll
        for (int j = 0; j < 4; ++j)
#pragma unroll
            for (int r = 0; r < 8; ++r) acc[i][j].e[r] = 0.0f;

    for (int k0 = 0; k0 < K; k0 += 32) {
        // async: TDM stages A tile (128 rows x 32 halves, padded to stride 40)
        if (wave == 0)
            tdm_load_2d(A + (size_t)bm * K + k0, As_lds,
                        /*tile_d0=*/32, /*tile_d1=*/128, /*stride0=*/(unsigned)K,
                        /*pad_interval 16 dw=*/3, /*pad 4 dw=*/3);
        // meanwhile: stage B tile transposed (read 8 n-contig halves, scatter along k)
#pragma unroll
        for (int j = 0; j < 2; ++j) {
            int c  = tid + 256 * j;
            int k  = c >> 4;                  // 0..31
            int n0 = (c & 15) << 3;           // 0..120
            uint4 pkt = *reinterpret_cast<const uint4*>(B + (size_t)(k0 + k) * N + bn + n0);
            const _Float16* hh = reinterpret_cast<const _Float16*>(&pkt);
#pragma unroll
            for (int i = 0; i < 8; ++i) Bs[(n0 + i) * 40 + k] = hh[i];
        }
        if (wave == 0) __builtin_amdgcn_s_wait_tensorcnt(0);
        __syncthreads();

        v16h af[2], bf[4];
#pragma unroll
        for (int i = 0; i < 2; ++i) af[i] = load_frag_a(&As[(wm * 32 + i * 16) * 40], 40);
#pragma unroll
        for (int j = 0; j < 4; ++j) bf[j] = load_frag_bT(&Bs[(wn * 64 + j * 16) * 40], 40);
#pragma unroll
        for (int i = 0; i < 2; ++i)
#pragma unroll
            for (int j = 0; j < 4; ++j)
                acc[i][j].v = __builtin_amdgcn_wmma_f32_16x16x32_f16(
                    false, af[i], false, bf[j], (short)0, acc[i][j].v, false, false);
        __syncthreads();
    }

    const int nn    = lane & 15;
    const int rbase = (lane >> 4) << 3;
#pragma unroll
    for (int i = 0; i < 2; ++i)
#pragma unroll
        for (int j = 0; j < 4; ++j) {
            int col = bn + wn * 64 + j * 16 + nn;
            float bv = (bias != nullptr) ? bias[col] : 0.0f;
#pragma unroll
            for (int r = 0; r < 8; ++r) {
                int row = bm + wm * 32 + i * 16 + rbase + r;
                float v = acc[i][j].e[r] + bv;
                if (RESID) v += resid[(size_t)row * N + col];
                if (RELU)  v = fmaxf(v, 0.0f);
                if (OUTF16) ((_Float16*)out)[(size_t)row * N + col] = (_Float16)v;
                else        ((float*)out)[(size_t)row * N + col]    = v;
            }
        }
}

// ---------------------------------------------------------------------------
// Flash attention, causal. Grid: (T/128, B*H). 8 waves; wave w owns q-rows
// [q0, q0+16). K tile staged by TDM (row-major, padded stride 72); V staged
// manually transposed [hd][key] for contiguous B fragments.
// ---------------------------------------------------------------------------
__global__ __launch_bounds__(256)
void attn_kernel(const _Float16* __restrict__ Q, const _Float16* __restrict__ Km,
                 const _Float16* __restrict__ Vm, _Float16* __restrict__ O,
                 int T, int Hn) {
    __shared__ __align__(16) _Float16 Kl[32 * 72];     // 32 keys x 64 hd (stride 72)
    __shared__ __align__(16) _Float16 Vt[64 * 40];     // 64 hd x 32 keys (stride 40)
    __shared__ __align__(16) _Float16 Pb[8 * 16 * 40]; // per-wave P 16x32 (stride 40)

    const int tid = threadIdx.x, lane = tid & 31, wave = tid >> 5;
    const int bh = blockIdx.y;
    const int b  = bh / Hn, h = bh % Hn;
    const int rowstride = Hn * 64;                     // = D halves
    const size_t headoff = (size_t)b * T * rowstride + (size_t)h * 64;
    const int q0 = blockIdx.x * 128 + wave * 16;
    const int nn  = lane & 15;
    const int hlf = lane >> 4;
    const unsigned Kl_lds = (unsigned)(size_t)&Kl[0];

    // load Q fragments (A layout, straight from global; contiguous in hd)
    v16h qf[2];
    {
        int m = lane & 15, kb = (lane >> 4) << 3;
        const _Float16* qrow = Q + headoff + (size_t)(q0 + m) * rowstride;
#pragma unroll
        for (int c = 0; c < 2; ++c) {
            AFrag f;
            const _Float16* p = qrow + c * 32 + kb;
#pragma unroll
            for (int i = 0; i < 8; ++i) { f.e[i] = p[i]; f.e[8 + i] = p[16 + i]; }
            qf[c] = f.v;
        }
    }

    float mrow[8], lrow[8];
    CFrag o[4];
#pragma unroll
    for (int r = 0; r < 8; ++r) { mrow[r] = -1e30f; lrow[r] = 0.0f; }
#pragma unroll
    for (int t = 0; t < 4; ++t)
#pragma unroll
        for (int r = 0; r < 8; ++r) o[t].e[r] = 0.0f;

    const int nk = (blockIdx.x + 1) * 128;             // causal: keys < q-block end
    for (int kb0 = 0; kb0 < nk; kb0 += 32) {
        // async: TDM stages K tile (32 keys x 64 halves, padded to stride 72)
        if (wave == 0)
            tdm_load_2d(Km + headoff + (size_t)kb0 * rowstride, Kl_lds,
                        /*tile_d0=*/64, /*tile_d1=*/32, /*stride0=*/(unsigned)rowstride,
                        /*pad_interval 32 dw=*/4, /*pad 4 dw=*/3);
        // meanwhile: stage V transposed
        {
            int row = tid >> 3;                        // key 0..31
            int col = (tid & 7) << 3;                  // hd chunk
            uint4 pkt = *reinterpret_cast<const uint4*>(
                Vm + headoff + (size_t)(kb0 + row) * rowstride + col);
            const _Float16* hh = reinterpret_cast<const _Float16*>(&pkt);
#pragma unroll
            for (int i = 0; i < 8; ++i) Vt[(col + i) * 40 + row] = hh[i];
        }
        if (wave == 0) __builtin_amdgcn_s_wait_tensorcnt(0);
        __syncthreads();

        // scores S[16q x 32k] = Q(16x64) @ K^T : 2 n-tiles x 2 k-chunks
        CFrag s[2];
#pragma unroll
        for (int j = 0; j < 2; ++j) {
#pragma unroll
            for (int r = 0; r < 8; ++r) s[j].e[r] = 0.0f;
            s[j].v = __builtin_amdgcn_wmma_f32_16x16x32_f16(
                false, qf[0], false, load_frag_bT(&Kl[(j * 16) * 72 + 0], 72),
                (short)0, s[j].v, false, false);
            s[j].v = __builtin_amdgcn_wmma_f32_16x16x32_f16(
                false, qf[1], false, load_frag_bT(&Kl[(j * 16) * 72 + 32], 72),
                (short)0, s[j].v, false, false);
        }

        // online softmax update (rows live across 16-lane halves, VALU reductions)
        float pj0[8], pj1[8];
#pragma unroll
        for (int r = 0; r < 8; ++r) {
            int row = q0 + (hlf << 3) + r;
            float v0 = s[0].e[r] * 0.125f;
            float v1 = s[1].e[r] * 0.125f;
            if (kb0 + nn > row)      v0 = -1e30f;
            if (kb0 + 16 + nn > row) v1 = -1e30f;
            float mb = rowmax16(fmaxf(v0, v1));
            float mo = mrow[r];
            float mn = fmaxf(mo, mb);
            float alpha = fast_exp(mo - mn);
            float p0 = fast_exp(v0 - mn);
            float p1 = fast_exp(v1 - mn);
            float rs = rowsum16(p0 + p1);
            lrow[r] = lrow[r] * alpha + rs;
            mrow[r] = mn;
            pj0[r] = p0; pj1[r] = p1;
#pragma unroll
            for (int t = 0; t < 4; ++t) o[t].e[r] *= alpha;
        }

        // P: C-layout -> A-layout via per-wave LDS scratch
        _Float16* Pw = &Pb[wave * 16 * 40];
#pragma unroll
        for (int r = 0; r < 8; ++r) {
            int prow = (hlf << 3) + r;
            Pw[prow * 40 + nn]      = (_Float16)pj0[r];
            Pw[prow * 40 + 16 + nn] = (_Float16)pj1[r];
        }
        asm volatile("s_wait_dscnt 0" ::: "memory");
        v16h pf = load_frag_a(Pw, 40);
#pragma unroll
        for (int t = 0; t < 4; ++t)
            o[t].v = __builtin_amdgcn_wmma_f32_16x16x32_f16(
                false, pf, false, load_frag_bT(&Vt[(t * 16) * 40], 40),
                (short)0, o[t].v, false, false);
        __syncthreads();
    }

    // normalize + store f16 [b, t, h, hd]
#pragma unroll
    for (int r = 0; r < 8; ++r) {
        float inv = 1.0f / lrow[r];
        int row = q0 + (hlf << 3) + r;
        _Float16* dst = O + headoff + (size_t)row * rowstride;
#pragma unroll
        for (int t = 0; t < 4; ++t)
            dst[t * 16 + nn] = (_Float16)(o[t].e[r] * inv);
    }
}

// ---------------------------------------------------------------------------
// Launch
// ---------------------------------------------------------------------------
extern "C" void kernel_launch(void* const* d_in, const int* in_sizes, int n_in,
                              void* d_out, int out_size, void* d_ws, size_t ws_size,
                              hipStream_t stream) {
    const float* x   = (const float*)d_in[0];
    const float* wq  = (const float*)d_in[1];
    const float* wk  = (const float*)d_in[2];
    const float* wv  = (const float*)d_in[3];
    const float* wo  = (const float*)d_in[4];
    const float* bo  = (const float*)d_in[5];
    const float* w1  = (const float*)d_in[6];
    const float* b1  = (const float*)d_in[7];
    const float* w2  = (const float*)d_in[8];
    const float* b2  = (const float*)d_in[9];
    const float* g1  = (const float*)d_in[10];
    const float* bl1 = (const float*)d_in[11];
    const float* g2  = (const float*)d_in[12];
    const float* bl2 = (const float*)d_in[13];
    float* out = (float*)d_out;

    const int D = 1024, Hh = 16, Bb = 4, T = 2048;
    const int M = Bb * T;        // 8192

    char*  ws  = (char*)d_ws;
    size_t off = 0;
    auto alloc = [&](size_t bytes) {
        char* p = ws + off;
        off += (bytes + 255) & ~(size_t)255;
        return p;
    };
    _Float16* wq16 = (_Float16*)alloc((size_t)D * D * 2);
    _Float16* wk16 = (_Float16*)alloc((size_t)D * D * 2);
    _Float16* wv16 = (_Float16*)alloc((size_t)D * D * 2);
    _Float16* wo16 = (_Float16*)alloc((size_t)D * D * 2);
    _Float16* w116 = (_Float16*)alloc((size_t)D * 4 * D * 2);
    _Float16* w216 = (_Float16*)alloc((size_t)D * 4 * D * 2);
    _Float16* h16  = (_Float16*)alloc((size_t)M * D * 2);     // ln1 out; reused for ln2 out
    _Float16* Q16  = (_Float16*)alloc((size_t)M * D * 2);
    _Float16* K16  = (_Float16*)alloc((size_t)M * D * 2);
    _Float16* V16  = (_Float16*)alloc((size_t)M * D * 2);
    _Float16* A16  = (_Float16*)alloc((size_t)M * D * 2);     // attn output
    float*    x1   = (float*)   alloc((size_t)M * D * 4);     // post-attn residual
    _Float16* U16  = Q16;  // 64MB FFN hidden aliases dead Q/K/V/attn region

    dim3 blk(256);
    int n = D * D;
    cvt_kernel<<<(n + 255) / 256, blk, 0, stream>>>(wq, wq16, n);
    cvt_kernel<<<(n + 255) / 256, blk, 0, stream>>>(wk, wk16, n);
    cvt_kernel<<<(n + 255) / 256, blk, 0, stream>>>(wv, wv16, n);
    cvt_kernel<<<(n + 255) / 256, blk, 0, stream>>>(wo, wo16, n);
    n = D * 4 * D;
    cvt_kernel<<<(n + 255) / 256, blk, 0, stream>>>(w1, w116, n);
    cvt_kernel<<<(n + 255) / 256, blk, 0, stream>>>(w2, w216, n);

    ln_kernel<<<M, blk, 0, stream>>>(x, g1, bl1, h16, D);

    dim3 gP(D / 128, M / 128);
    gemm_f16_kernel<false, false, true><<<gP, blk, 0, stream>>>(h16, wq16, nullptr, nullptr, Q16, M, D, D);
    gemm_f16_kernel<false, false, true><<<gP, blk, 0, stream>>>(h16, wk16, nullptr, nullptr, K16, M, D, D);
    gemm_f16_kernel<false, false, true><<<gP, blk, 0, stream>>>(h16, wv16, nullptr, nullptr, V16, M, D, D);

    dim3 gA(T / 128, Bb * Hh);
    attn_kernel<<<gA, blk, 0, stream>>>(Q16, K16, V16, A16, T, Hh);

    gemm_f16_kernel<false, true, false><<<gP, blk, 0, stream>>>(A16, wo16, bo, x, x1, M, D, D);

    ln_kernel<<<M, blk, 0, stream>>>(x1, g2, bl2, h16, D);

    dim3 gF(4 * D / 128, M / 128);
    gemm_f16_kernel<true,  false, true ><<<gF, blk, 0, stream>>>(h16, w116, b1, nullptr, U16, M, 4 * D, D);
    gemm_f16_kernel<false, true,  false><<<gP, blk, 0, stream>>>(U16, w216, b2, x1, out, M, D, 4 * D);
}
